// NeuronDecoder_51187420234199
// MI455X (gfx1250) — matrix operand order repaired and checked by reading, not compiled, not run on hardware
//
#include <hip/hip_runtime.h>

typedef __attribute__((ext_vector_type(16))) _Float16 v16h;
typedef __attribute__((ext_vector_type(8)))  float    v8f;
typedef int v4i_ __attribute__((vector_size(16)));

#define HIDDEN  4096
#define MROWS   2048      // B*S = 4*512
#define NKEYS   512
#define NBH     64        // B*HEADS
#define VEC     64
#define SSTRIDE 520       // f32 slots per score row in LDS (520 % 64 == 8)
#define LDPAD   40        // f16 per staged tile row: 32 + 8 pad (80B, 16B aligned, bank-clean)

// ---------------------------------------------------------------- CDNA5 async copy

#if __has_builtin(__builtin_amdgcn_global_load_async_to_lds_b128)
#define HAVE_ASYNC 1
#else
#define HAVE_ASYNC 0
#endif

__device__ __forceinline__ void stage16(const _Float16* g, _Float16* l) {
#if HAVE_ASYNC
  __builtin_amdgcn_global_load_async_to_lds_b128(
      (v4i_*)g, (v4i_*)l, /*offset=*/0, /*cpol=*/0);
#else
  *(float4*)l = *(const float4*)g;   // sync fallback staging
#endif
}

__device__ __forceinline__ void wait_async() {
#if HAVE_ASYNC
#if __has_builtin(__builtin_amdgcn_s_wait_asynccnt)
  __builtin_amdgcn_s_wait_asynccnt(0);
#else
  asm volatile("s_wait_asynccnt 0" ::: "memory");
#endif
#endif
}

// ---------------------------------------------------------------- WMMA helpers

__device__ __forceinline__ v8f wmma16(v16h a, v16h b, v8f c) {
  // D = A(16x32 f16) * B(32x16 f16) + C(16x16 f32)
  return __builtin_amdgcn_wmma_f32_16x16x32_f16(
      /*neg_a=*/false, a, /*neg_b=*/false, b,
      /*c_mod=*/(short)0, c, /*reuse_a=*/false, /*reuse_b=*/false);
}

// A-matrix 16x32 fragment, source row-major [row][k], leading dim ld.
// ISA layout: lanes 0-15 (M=lane)  -> elems 0..7 = K 0..7,  elems 8..15 = K 16..23
//             lanes 16-31 (M=l-16) -> elems 0..7 = K 8..15, elems 8..15 = K 24..31
__device__ __forceinline__ v16h load_fragA(const _Float16* __restrict__ tile,
                                           int ld, int lane) {
  int r  = lane & 15;
  int ko = (lane & 16) ? 8 : 0;
  const _Float16* p = tile + (size_t)r * ld + ko;
  v16h f;
#pragma unroll
  for (int i = 0; i < 8; ++i) { f[i] = p[i]; f[i + 8] = p[i + 16]; }
  return f;
}

// B-matrix 32x16 fragment, source stored as [col][k] (column-major along K), ld over k.
// ISA layout: lanes 0-15 (N=lane)  -> elems 0..7 = K 0..7,  elems 8..15 = K 16..23
//             lanes 16-31 (N=l-16) -> elems 0..7 = K 8..15, elems 8..15 = K 24..31
__device__ __forceinline__ v16h load_fragB(const _Float16* __restrict__ tile,
                                           int ld, int lane) {
  int c  = lane & 15;
  int ko = (lane & 16) ? 8 : 0;
  const _Float16* p = tile + (size_t)c * ld + ko;
  v16h f;
#pragma unroll
  for (int i = 0; i < 8; ++i) { f[i] = p[i]; f[i + 8] = p[i + 16]; }
  return f;
}

// ---------------------------------------------------------------- conversions

__global__ void cvt_f32_f16(const float* __restrict__ in, _Float16* __restrict__ out, int n) {
  for (int i = blockIdx.x * blockDim.x + threadIdx.x; i < n; i += gridDim.x * blockDim.x)
    out[i] = (_Float16)in[i];
}

// keys:   (B,N,1024) -> K[bh][n][v]  = keys[b][n][v*16+h]    (B operand for Q*K^T)
// values: (B,N,1024) -> V[bh][vv][n] = values[b][n][vv*16+h] (B operand for P*V)
__global__ void cvt_kv(const float* __restrict__ keys, const float* __restrict__ values,
                       _Float16* __restrict__ kt, _Float16* __restrict__ vt, int n) {
  for (int i = blockIdx.x * blockDim.x + threadIdx.x; i < n; i += gridDim.x * blockDim.x) {
    int j  = i & 1023;
    int nn = (i >> 10) & 511;
    int b  = i >> 19;              // / (512*1024)
    int h  = j & 15, v = j >> 4;
    int bh = b * 16 + h;
    kt[((size_t)bh * NKEYS + nn) * VEC + v]  = (_Float16)keys[i];
    vt[((size_t)bh * VEC + v) * NKEYS + nn]  = (_Float16)values[i];
  }
}

// ---------------------------------------------------------------- big GEMM
// out(row,col) = sum_k A[row][k] * B[col][k] + bias[col]
// 64x256 block tile, 8 waves of 32x64, double-buffered async global->LDS staging.
// MODE 0: scatter f16 into q_heads[bh][m][v] (fused reshape/transpose)
// MODE 1: write f32 row-major to d_out

template <int MODE>
__global__ void __launch_bounds__(256)
gemm_wmma(const _Float16* __restrict__ A, const _Float16* __restrict__ Bm,
          const float* __restrict__ bias, void* __restrict__ outp) {
  const int K = HIDDEN;
  __shared__ _Float16 As[2][64 * LDPAD];
  __shared__ _Float16 Bs[2][256 * LDPAD];

  int t    = threadIdx.x;
  int lane = t & 31;
  int wave = t >> 5;
  int mBlk0 = blockIdx.y * 64;
  int nBlk0 = blockIdx.x * 256;
  int wm = (wave & 1) * 32;      // wave's rows within block tile
  int wn = (wave >> 1) * 64;     // wave's cols within block tile

  // staging geometry: b128 chunks of 8 halfs
  int arow = t >> 2, aseg = (t & 3) * 8;             // A: 64 rows x 4 segs = 256 chunks
  auto stage = [&](int buf, int k0) {
    stage16(A + (size_t)(mBlk0 + arow) * K + k0 + aseg, &As[buf][arow * LDPAD + aseg]);
#pragma unroll
    for (int c = 0; c < 4; ++c) {                    // B: 256 rows x 4 segs = 1024 chunks
      int idx = t + c * 256;
      int row = idx >> 2, seg = (idx & 3) * 8;
      stage16(Bm + (size_t)(nBlk0 + row) * K + k0 + seg, &Bs[buf][row * LDPAD + seg]);
    }
  };

  v8f zero = {};
  v8f acc[2][4];
#pragma unroll
  for (int mt = 0; mt < 2; ++mt)
#pragma unroll
    for (int nt = 0; nt < 4; ++nt) acc[mt][nt] = zero;

  stage(0, 0);
  const int KT = K / 32;
  for (int kt = 0; kt < KT; ++kt) {
    wait_async();
    __syncthreads();                       // buffer kt&1 published to all waves
    if (kt + 1 < KT) stage((kt + 1) & 1, (kt + 1) * 32);

    const _Float16* Ab = &As[kt & 1][wm * LDPAD];
    const _Float16* Bb = &Bs[kt & 1][wn * LDPAD];
    v16h a0 = load_fragA(Ab, LDPAD, lane);
    v16h a1 = load_fragA(Ab + 16 * LDPAD, LDPAD, lane);
    v16h b[4];
#pragma unroll
    for (int nt = 0; nt < 4; ++nt) b[nt] = load_fragB(Bb + nt * 16 * LDPAD, LDPAD, lane);
#pragma unroll
    for (int nt = 0; nt < 4; ++nt) {
      acc[0][nt] = wmma16(a0, b[nt], acc[0][nt]);
      acc[1][nt] = wmma16(a1, b[nt], acc[1][nt]);
    }
    __syncthreads();                       // reads done before buffer is restaged
  }

  int rlo = ((lane >> 4) & 1) * 8;         // C/D: elem i -> row i + 8*(lane/16)
  int cl  = lane & 15;
#pragma unroll
  for (int mt = 0; mt < 2; ++mt)
#pragma unroll
    for (int nt = 0; nt < 4; ++nt)
#pragma unroll
      for (int i = 0; i < 8; ++i) {
        int row = mBlk0 + wm + mt * 16 + rlo + i;
        int col = nBlk0 + wn + nt * 16 + cl;
        float val = acc[mt][nt][i] + bias[col];
        if (MODE == 0) {
          int b = row >> 9, s = row & 511;
          int d = col >> 10, rem = col & 1023;
          int v = rem >> 4, h = rem & 15;
          ((_Float16*)outp)[(((size_t)(b * 16 + h) * MROWS) + (s * 4 + d)) * VEC + v] =
              (_Float16)val;
        } else {
          ((float*)outp)[(size_t)row * HIDDEN + col] = val;
        }
      }
}

// ---------------------------------------------------------------- fused attention
// 4 waves / block, each wave owns one 16-row strip of one (b,h):
// scores(16x512) -> LDS -> softmax (in place, f16) -> P @ V (16x64) -> attn layout

__global__ void attn_wmma(const _Float16* __restrict__ qh, const _Float16* __restrict__ kt,
                          const _Float16* __restrict__ vt, float* __restrict__ attn) {
  extern __shared__ float lds[];
  int lane = threadIdx.x & 31;
  int wave = threadIdx.x >> 5;
  float* sc   = lds + (size_t)wave * 16 * SSTRIDE;
  float* rinv = lds + (size_t)4 * 16 * SSTRIDE + wave * 16;

  int strip = blockIdx.x * 4 + wave;   // 0..8191
  int bh = strip >> 7;                 // 128 strips per (b,h)
  int ms = strip & 127;
  const _Float16* Q  = qh + ((size_t)bh * MROWS + ms * 16) * VEC;
  const _Float16* Kt = kt + (size_t)bh * NKEYS * VEC;   // [n][v]
  const _Float16* Vt = vt + (size_t)bh * VEC * NKEYS;   // [vv][n]

  int rlo = ((lane >> 4) & 1) * 8;
  int cl  = lane & 15;
  v8f zero = {};

  // ---- phase 1: scores = Q(16x64) * K^T(64x512) into LDS (f32)
  v16h a0 = load_fragA(Q,      VEC, lane);   // k = 0..31
  v16h a1 = load_fragA(Q + 32, VEC, lane);   // k = 32..63
  for (int nt = 0; nt < 32; ++nt) {
    v16h b0 = load_fragB(Kt + (size_t)nt * 16 * VEC,      VEC, lane);
    v16h b1 = load_fragB(Kt + (size_t)nt * 16 * VEC + 32, VEC, lane);
    v8f s = zero;
    s = wmma16(a0, b0, s);
    s = wmma16(a1, b1, s);
#pragma unroll
    for (int i = 0; i < 8; ++i)
      sc[(rlo + i) * SSTRIDE + nt * 16 + cl] = s[i];
  }
  asm volatile("s_wait_dscnt 0" ::: "memory");

  // ---- phase 2: per-row softmax (scale 1/8), probs stored in place as f16
  if (lane < 16) {
    float* row = sc + lane * SSTRIDE;
    float mx = -3.4e38f;
    for (int j = 0; j < 512; ++j) mx = fmaxf(mx, row[j]);
    mx *= 0.125f;
    float ssum = 0.f;
    for (int j = 0; j < 512; ++j) {
      float e = __expf(row[j] * 0.125f - mx);
      ssum += e;
      *(_Float16*)(row + j) = (_Float16)e;   // low half of the f32 slot
    }
    rinv[lane] = 1.f / ssum;
  }
  asm volatile("s_wait_dscnt 0" ::: "memory");

  // ---- phase 3: out(16x64) = P(16x512) * V(512x64), row-scaled by 1/sum
  v8f o[4];
#pragma unroll
  for (int ct = 0; ct < 4; ++ct) o[ct] = zero;
  const _Float16* ph = (const _Float16*)sc;
  int r  = lane & 15;
  int ko = (lane & 16) ? 8 : 0;
  for (int ks = 0; ks < 512; ks += 32) {
    v16h ap;
    int base = r * SSTRIDE + ks + ko;
#pragma unroll
    for (int i = 0; i < 8; ++i) {
      ap[i]     = ph[2 * (base + i)];
      ap[i + 8] = ph[2 * (base + 16 + i)];
    }
#pragma unroll
    for (int ct = 0; ct < 4; ++ct)
      o[ct] = wmma16(ap, load_fragB(Vt + (size_t)ct * 16 * NKEYS + ks, NKEYS, lane), o[ct]);
  }

  // ---- epilogue: fused reshape/transpose into (B,S,HIDDEN) f32
  int b = bh >> 4, h = bh & 15;
#pragma unroll
  for (int ct = 0; ct < 4; ++ct)
#pragma unroll
    for (int i = 0; i < 8; ++i) {
      int rowl = rlo + i;
      int m = ms * 16 + rowl;
      int s = m >> 2, d = m & 3;
      int vv = ct * 16 + cl;
      float val = o[ct][i] * rinv[rowl];
      attn[(size_t)(b * 512 + s) * HIDDEN + h * 256 + d * 64 + vv] = val;
    }
}

// ---------------------------------------------------------------- LayerNorm

__global__ void ln_kernel(const float* __restrict__ x, const float* __restrict__ g,
                          const float* __restrict__ bb, _Float16* __restrict__ y) {
  __shared__ float red[512];
  int r = blockIdx.x;
  const float* row = x + (size_t)r * HIDDEN;
  float s = 0.f, s2 = 0.f;
  for (int j = threadIdx.x; j < HIDDEN; j += 256) {
    float v = row[j]; s += v; s2 += v * v;
  }
  red[threadIdx.x] = s; red[256 + threadIdx.x] = s2;
  __syncthreads();
  for (int off = 128; off > 0; off >>= 1) {
    if ((int)threadIdx.x < off) {
      red[threadIdx.x]       += red[threadIdx.x + off];
      red[256 + threadIdx.x] += red[256 + threadIdx.x + off];
    }
    __syncthreads();
  }
  float mu  = red[0] * (1.f / HIDDEN);
  float var = red[256] * (1.f / HIDDEN) - mu * mu;
  float inv = rsqrtf(var + 1e-12f);
  for (int j = threadIdx.x; j < HIDDEN; j += 256) {
    float v = (row[j] - mu) * inv * g[j] + bb[j];
    y[(size_t)r * HIDDEN + j] = (_Float16)v;
  }
}

// ---------------------------------------------------------------- launch

extern "C" void kernel_launch(void* const* d_in, const int* in_sizes, int n_in,
                              void* d_out, int out_size, void* d_ws, size_t ws_size,
                              hipStream_t stream) {
  (void)in_sizes; (void)n_in; (void)out_size; (void)ws_size;
  const float* emb  = (const float*)d_in[0];
  const float* keys = (const float*)d_in[1];
  const float* vals = (const float*)d_in[2];
  const float* Wq   = (const float*)d_in[3];
  const float* bq   = (const float*)d_in[4];
  const float* Wre  = (const float*)d_in[5];
  const float* bre  = (const float*)d_in[6];
  const float* lng  = (const float*)d_in[7];
  const float* lnb  = (const float*)d_in[8];
  float* out = (float*)d_out;

  char* p = (char*)d_ws;
  auto take = [&](size_t bytes) {
    char* r = p; p += (bytes + 255) & ~(size_t)255; return r;
  };
  _Float16* Wq_h  = (_Float16*)take((size_t)HIDDEN * HIDDEN * 2);
  _Float16* Wre_h = (_Float16*)take((size_t)HIDDEN * HIDDEN * 2);
  _Float16* emb_h = (_Float16*)take((size_t)MROWS * HIDDEN * 2);
  _Float16* q_h   = (_Float16*)take((size_t)NBH * MROWS * VEC * 2);
  _Float16* kt_h  = (_Float16*)take((size_t)NBH * NKEYS * VEC * 2);
  _Float16* vt_h  = (_Float16*)take((size_t)NBH * VEC * NKEYS * 2);
  float*    attn  = (float*)   take((size_t)MROWS * HIDDEN * 4);
  _Float16* ln_h  = (_Float16*)take((size_t)MROWS * HIDDEN * 2);

  cvt_f32_f16<<<4096, 256, 0, stream>>>(Wq,  Wq_h,  HIDDEN * HIDDEN);
  cvt_f32_f16<<<4096, 256, 0, stream>>>(Wre, Wre_h, HIDDEN * HIDDEN);
  cvt_f32_f16<<<2048, 256, 0, stream>>>(emb, emb_h, MROWS * HIDDEN);
  cvt_kv<<<2048, 256, 0, stream>>>(keys, vals, kt_h, vt_h, 4 * NKEYS * 1024);

  dim3 gg(HIDDEN / 256, MROWS / 64);
  gemm_wmma<0><<<gg, 256, 0, stream>>>(emb_h, Wq_h, bq, (void*)q_h);

  size_t ldsB = (size_t)4 * 16 * SSTRIDE * 4 + 4 * 16 * 4;  // scores + rowsum
  attn_wmma<<<(NBH * 128) / 4, 128, ldsB, stream>>>(q_h, kt_h, vt_h, attn);

  ln_kernel<<<MROWS, 256, 0, stream>>>(attn, lng, lnb, ln_h);

  gemm_wmma<1><<<gg, 256, 0, stream>>>(ln_h, Wre_h, bre, (void*)out);
}
